// AttnBlock_51866025066997
// MI455X (gfx1250) — matrix operand order repaired and checked by reading, not compiled, not run on hardware
//
#include <hip/hip_runtime.h>
#include <hip/hip_bf16.h>

typedef unsigned short u16;
typedef unsigned int u32;
typedef __attribute__((ext_vector_type(16))) __bf16 v16bf;
typedef __attribute__((ext_vector_type(8)))  float  v8f;
typedef int v4i __attribute__((vector_size(16)));

#define NTOK 4096
#define CDIM 512

union BF16x16 { v16bf v; u16 u[16]; uint4 q[2]; };
union BF16x8  { u16 u[8]; uint4 q4; };

__device__ __forceinline__ u16 f2bf(float f) {
    u32 u = __float_as_uint(f);
    return (u16)((u + 0x7FFFu + ((u >> 16) & 1u)) >> 16);
}

// A-matrix fragment, 16x32 bf16 (M x K), from row-major src (ld in elements).
__device__ __forceinline__ v16bf load_frag_A(const u16* base, int ld, int lane) {
    int m = lane & 15, h = (lane >> 4) & 1;
    const u16* p = base + (size_t)m * ld + h * 16;
    BF16x16 r;
    r.q[0] = *(const uint4*)(p);
    r.q[1] = *(const uint4*)(p + 8);
    return r.v;
}

// B-matrix fragment, 32x16 bf16 (K x N), where B(k,n) = X[n][k], X row-major.
__device__ __forceinline__ v16bf load_frag_B(const u16* base, int ld, int lane) {
    int n = lane & 15, h = (lane >> 4) & 1;
    const u16* p = base + (size_t)n * ld + h * 16;
    BF16x16 r;
    r.q[0] = *(const uint4*)(p);
    r.q[1] = *(const uint4*)(p + 8);
    return r.v;
}

__device__ __forceinline__ v8f wmma_bf16(v16bf a, v16bf b, v8f c) {
    return __builtin_amdgcn_wmma_f32_16x16x32_bf16(false, a, false, b, (short)0, c, false, false);
}

// ---- CDNA5 async global->LDS copy (16B per lane), with portable fallback ----
__device__ __forceinline__ void async_cp16(const u16* g, u16* l) {
#if __has_builtin(__builtin_amdgcn_global_load_async_to_lds_b128)
    __builtin_amdgcn_global_load_async_to_lds_b128(
        (__attribute__((address_space(1))) v4i*)g,
        (__attribute__((address_space(3))) v4i*)l, 0, 0);
#else
    *(uint4*)l = *(const uint4*)g;
#endif
}

template<int N>
__device__ __forceinline__ void wait_asynccnt() {
#if __has_builtin(__builtin_amdgcn_s_wait_asynccnt)
    __builtin_amdgcn_s_wait_asynccnt(N);
#else
    asm volatile("s_wait_asynccnt %0" :: "i"(N) : "memory");
#endif
}

// ---------------- weight f32 -> bf16 ----------------
__global__ void cvtw_kernel(const float* __restrict__ W, u16* __restrict__ out, int n) {
    int i = blockIdx.x * blockDim.x + threadIdx.x;
    if (i < n) out[i] = f2bf(W[i]);
}

// ---------------- GroupNorm: x[B,C,N] -> xn[B,N,C] bf16 ----------------
__global__ void gn_kernel(const float* __restrict__ x, const float* __restrict__ gamma,
                          const float* __restrict__ beta, u16* __restrict__ xn) {
    int bg = blockIdx.x; int b = bg >> 5; int g = bg & 31;
    const float* xp = x + ((size_t)b * CDIM + g * 16) * NTOK;   // [16][4096]
    float s = 0.f, ss = 0.f;
    for (int i = threadIdx.x; i < 16 * NTOK; i += blockDim.x) {
        float v = xp[i]; s += v; ss += v * v;
    }
    __shared__ float rs[256], rss[256];
    rs[threadIdx.x] = s; rss[threadIdx.x] = ss; __syncthreads();
    for (int off = 128; off > 0; off >>= 1) {
        if ((int)threadIdx.x < off) { rs[threadIdx.x] += rs[threadIdx.x + off];
                                      rss[threadIdx.x] += rss[threadIdx.x + off]; }
        __syncthreads();
    }
    float inv = 1.f / (16.f * NTOK);
    float mean = rs[0] * inv;
    float var  = rss[0] * inv - mean * mean;
    float rstd = rsqrtf(var + 1e-5f);
    __shared__ float sc[16], sh[16];
    if (threadIdx.x < 16) {
        float gm = gamma[g * 16 + threadIdx.x], bt = beta[g * 16 + threadIdx.x];
        sc[threadIdx.x] = gm * rstd;
        sh[threadIdx.x] = bt - mean * rstd * gm;
    }
    __syncthreads();
    for (int tok = threadIdx.x; tok < NTOK; tok += blockDim.x) {
        BF16x16 o;
        #pragma unroll
        for (int c = 0; c < 16; c++)
            o.u[c] = f2bf(xp[(size_t)c * NTOK + tok] * sc[c] + sh[c]);
        uint4* dst = (uint4*)(xn + ((size_t)b * NTOK + tok) * CDIM + g * 16);
        dst[0] = o.q[0]; dst[1] = o.q[1];
    }
}

// ------------- Q/K projection: Y[b,n,c] = (xn @ W^T + bias)*scale, bf16 -------------
// one wave per block: 16 tokens x 64 couts; k-loop software pipelined
__global__ void proj_nk_kernel(const u16* __restrict__ X, const u16* __restrict__ W,
                               const float* __restrict__ bias, float scale,
                               u16* __restrict__ Y) {
    int lane = threadIdx.x & 31;
    int n0 = blockIdx.x * 16, c0 = blockIdx.y * 64, b = blockIdx.z;
    const u16* Xb = X + (size_t)b * NTOK * CDIM + (size_t)n0 * CDIM;
    v8f zero = {0,0,0,0,0,0,0,0};
    v8f acc[4] = {zero, zero, zero, zero};
    v16bf a = load_frag_A(Xb, CDIM, lane);
    v16bf bb[4];
    #pragma unroll
    for (int t = 0; t < 4; t++) bb[t] = load_frag_B(W + (size_t)(c0 + t * 16) * CDIM, CDIM, lane);
    #pragma unroll
    for (int k0 = 0; k0 < CDIM; k0 += 32) {
        v16bf an = a; v16bf bn[4] = {bb[0], bb[1], bb[2], bb[3]};
        if (k0 + 32 < CDIM) {
            an = load_frag_A(Xb + k0 + 32, CDIM, lane);
            #pragma unroll
            for (int t = 0; t < 4; t++)
                bn[t] = load_frag_B(W + (size_t)(c0 + t * 16) * CDIM + k0 + 32, CDIM, lane);
        }
        #pragma unroll
        for (int t = 0; t < 4; t++) acc[t] = wmma_bf16(a, bb[t], acc[t]);
        a = an;
        #pragma unroll
        for (int t = 0; t < 4; t++) bb[t] = bn[t];
    }
    int nn = lane & 15, h = (lane >> 4) & 1;
    #pragma unroll
    for (int t = 0; t < 4; t++) {
        float bv = bias[c0 + t * 16 + nn];
        #pragma unroll
        for (int r = 0; r < 8; r++) {
            int tok = n0 + r + 8 * h;
            Y[(size_t)b * NTOK * CDIM + (size_t)tok * CDIM + c0 + t * 16 + nn] =
                f2bf((acc[t][r] + bv) * scale);
        }
    }
}

// ------------- V projection, transposed output: VT[b,c,n] bf16 -------------
__global__ void proj_v_kernel(const u16* __restrict__ X, const u16* __restrict__ W,
                              const float* __restrict__ bias, u16* __restrict__ VT) {
    int lane = threadIdx.x & 31;
    int c0 = blockIdx.x * 16, n0 = blockIdx.y * 64, b = blockIdx.z;
    const u16* Xb = X + (size_t)b * NTOK * CDIM;
    v8f zero = {0,0,0,0,0,0,0,0};
    v8f acc[4] = {zero, zero, zero, zero};
    v16bf a = load_frag_A(W + (size_t)c0 * CDIM, CDIM, lane);
    v16bf bb[4];
    #pragma unroll
    for (int t = 0; t < 4; t++) bb[t] = load_frag_B(Xb + (size_t)(n0 + t * 16) * CDIM, CDIM, lane);
    #pragma unroll
    for (int k0 = 0; k0 < CDIM; k0 += 32) {
        v16bf an = a; v16bf bn[4] = {bb[0], bb[1], bb[2], bb[3]};
        if (k0 + 32 < CDIM) {
            an = load_frag_A(W + (size_t)c0 * CDIM + k0 + 32, CDIM, lane);
            #pragma unroll
            for (int t = 0; t < 4; t++)
                bn[t] = load_frag_B(Xb + (size_t)(n0 + t * 16) * CDIM + k0 + 32, CDIM, lane);
        }
        #pragma unroll
        for (int t = 0; t < 4; t++) acc[t] = wmma_bf16(a, bb[t], acc[t]);
        a = an;
        #pragma unroll
        for (int t = 0; t < 4; t++) bb[t] = bn[t];
    }
    int nn = lane & 15, h = (lane >> 4) & 1;
    #pragma unroll
    for (int t = 0; t < 4; t++)
        #pragma unroll
        for (int r = 0; r < 8; r++) {
            int cout = c0 + r + 8 * h;
            VT[(size_t)b * CDIM * NTOK + (size_t)cout * NTOK + n0 + t * 16 + nn] =
                f2bf(acc[t][r] + bias[cout]);
        }
}

// ------------- fused flash attention with async-LDS K/V staging -------------
// Workgroup = 4 waves, each owns 16 query rows; K/V 32-key blocks staged in LDS
// (double buffered) via GLOBAL_LOAD_ASYNC_TO_LDS_B128, shared by all 4 waves.
// Computes S^T = K Q^T (query index in lane dim -> per-lane softmax stats),
// then O^T = V^T P^T.
__global__ void __launch_bounds__(128, 1)
flash_kernel(const u16* __restrict__ Q, const u16* __restrict__ K,
             const u16* __restrict__ VT, u16* __restrict__ O) {
    __shared__ u16 Ks[2][32 * CDIM];   // 2 x 32KB : keys x channels (row-major, contiguous)
    __shared__ u16 Vs[2][CDIM * 32];   // 2 x 32KB : channels x keys
    int tid = threadIdx.x;
    int lane = tid & 31, wave = tid >> 5;
    int q0 = (blockIdx.x * 4 + wave) * 16, b = blockIdx.y;
    const u16* Qb = Q + (size_t)b * NTOK * CDIM + (size_t)q0 * CDIM;
    const u16* Kb = K + (size_t)b * NTOK * CDIM;
    const u16* Vb = VT + (size_t)b * CDIM * NTOK;
    int h = (lane >> 4) & 1;

    v16bf qf[16];
    #pragma unroll
    for (int cc = 0; cc < 16; cc++) qf[cc] = load_frag_B(Qb + cc * 32, CDIM, lane);

    v8f zero = {0,0,0,0,0,0,0,0};
    v8f o[32];
    #pragma unroll
    for (int i = 0; i < 32; i++) o[i] = zero;
    float m = -1e30f, l = 0.f;

    // stage one 32-key block (K: contiguous 32KB; V: 512 rows x 64B) into LDS buf
    auto issue = [&](int kb, int buf) {
        const u16* ksrc = Kb + (size_t)kb * CDIM;
        #pragma unroll
        for (int i = 0; i < 16; i++) {                 // 2048 x 16B chunks / 128 thr
            int ch = tid + i * 128;
            async_cp16(ksrc + ch * 8, &Ks[buf][ch * 8]);
        }
        #pragma unroll
        for (int i = 0; i < 16; i++) {
            int ch = tid + i * 128;
            int row = ch >> 2, off = (ch & 3) * 8;     // 4 x 16B per channel row
            async_cp16(Vb + (size_t)row * NTOK + kb + off, &Vs[buf][row * 32 + off]);
        }
    };

    issue(0, 0);
    for (int ib = 0; ib < NTOK / 32; ib++) {
        int buf = ib & 1;
        if (ib + 1 < NTOK / 32) {
            issue(ib + 1, buf ^ 1);
            wait_asynccnt<32>();    // in-order completion: current block is resident
        } else {
            wait_asynccnt<0>();
        }
        __syncthreads();

        const u16* Kl = &Ks[buf][0];
        const u16* Vl = &Vs[buf][0];
        v8f sa = zero, sb = zero;     // S^T tiles: keys +0..15 / +16..31
        #pragma unroll
        for (int cc = 0; cc < 16; cc++) {
            v16bf ka  = load_frag_A(Kl + cc * 32, CDIM, lane);
            v16bf ka2 = load_frag_A(Kl + 16 * CDIM + cc * 32, CDIM, lane);
            sa = wmma_bf16(ka,  qf[cc], sa);
            sb = wmma_bf16(ka2, qf[cc], sb);
        }
        // per-lane (= per-query) online softmax
        float t = sa[0];
        #pragma unroll
        for (int r = 1; r < 8; r++) t = fmaxf(t, sa[r]);
        #pragma unroll
        for (int r = 0; r < 8; r++) t = fmaxf(t, sb[r]);
        t = fmaxf(t, __shfl_xor(t, 16, 32));
        float mnew = fmaxf(m, t);
        float f = __expf(m - mnew);
        float pa[8], pb[8], s = 0.f;
        #pragma unroll
        for (int r = 0; r < 8; r++) {
            pa[r] = __expf(sa[r] - mnew);
            pb[r] = __expf(sb[r] - mnew);
            s += pa[r] + pb[r];
        }
        s += __shfl_xor(s, 16, 32);
        l = l * f + s;
        m = mnew;
        #pragma unroll
        for (int i = 0; i < 32; i++)
            #pragma unroll
            for (int r = 0; r < 8; r++) o[i][r] *= f;
        // pack P^T (D layout) into B fragment: lane half h needs keys 16h+j
        BF16x16 pf;
        #pragma unroll
        for (int r = 0; r < 8; r++) {
            float sel  = h ? pa[r] : pb[r];
            float recv = __shfl_xor(sel, 16, 32);
            float lo = h ? recv  : pa[r];
            float hi = h ? pb[r] : recv;
            pf.u[r]     = f2bf(lo);
            pf.u[8 + r] = f2bf(hi);
        }
        #pragma unroll
        for (int cv = 0; cv < 32; cv++) {
            v16bf vf = load_frag_A(Vl + (size_t)(cv * 16) * 32, 32, lane);
            o[cv] = wmma_bf16(vf, pf.v, o[cv]);
        }
        __syncthreads();
    }
    float linv = 1.f / l;
    int nn = lane & 15;
    u16* Ob = O + (size_t)b * NTOK * CDIM + (size_t)(q0 + nn) * CDIM + 8 * h;
    #pragma unroll
    for (int cv = 0; cv < 32; cv++) {
        BF16x8 pk;
        #pragma unroll
        for (int r = 0; r < 8; r++) pk.u[r] = f2bf(o[cv][r] * linv);
        *(uint4*)(Ob + cv * 16) = pk.q4;
    }
}

// ------------- output projection + bias + residual, f32 [B,C,N] -------------
__global__ void proj_out_kernel(const u16* __restrict__ A, const u16* __restrict__ W,
                                const float* __restrict__ bias, const float* __restrict__ xres,
                                float* __restrict__ out) {
    int lane = threadIdx.x & 31;
    int c0 = blockIdx.x * 16, n0 = blockIdx.y * 64, b = blockIdx.z;
    const u16* Ab = A + (size_t)b * NTOK * CDIM;
    v8f zero = {0,0,0,0,0,0,0,0};
    v8f acc[4] = {zero, zero, zero, zero};
    v16bf a = load_frag_A(W + (size_t)c0 * CDIM, CDIM, lane);
    v16bf bb[4];
    #pragma unroll
    for (int t = 0; t < 4; t++) bb[t] = load_frag_B(Ab + (size_t)(n0 + t * 16) * CDIM, CDIM, lane);
    #pragma unroll
    for (int k0 = 0; k0 < CDIM; k0 += 32) {
        v16bf an = a; v16bf bn[4] = {bb[0], bb[1], bb[2], bb[3]};
        if (k0 + 32 < CDIM) {
            an = load_frag_A(W + (size_t)c0 * CDIM + k0 + 32, CDIM, lane);
            #pragma unroll
            for (int t = 0; t < 4; t++)
                bn[t] = load_frag_B(Ab + (size_t)(n0 + t * 16) * CDIM + k0 + 32, CDIM, lane);
        }
        #pragma unroll
        for (int t = 0; t < 4; t++) acc[t] = wmma_bf16(a, bb[t], acc[t]);
        a = an;
        #pragma unroll
        for (int t = 0; t < 4; t++) bb[t] = bn[t];
    }
    int nn = lane & 15, h = (lane >> 4) & 1;
    #pragma unroll
    for (int t = 0; t < 4; t++)
        #pragma unroll
        for (int r = 0; r < 8; r++) {
            int cout = c0 + r + 8 * h;
            size_t idx = (size_t)b * CDIM * NTOK + (size_t)cout * NTOK + n0 + t * 16 + nn;
            out[idx] = acc[t][r] + bias[cout] + xres[idx];
        }
}

extern "C" void kernel_launch(void* const* d_in, const int* in_sizes, int n_in,
                              void* d_out, int out_size, void* d_ws, size_t ws_size,
                              hipStream_t stream) {
    const float* x    = (const float*)d_in[0];
    const float* gn_w = (const float*)d_in[1];
    const float* gn_b = (const float*)d_in[2];
    const float* Wq   = (const float*)d_in[3];
    const float* bq   = (const float*)d_in[4];
    const float* Wk   = (const float*)d_in[5];
    const float* bk   = (const float*)d_in[6];
    const float* Wv   = (const float*)d_in[7];
    const float* bv   = (const float*)d_in[8];
    const float* Wo   = (const float*)d_in[9];
    const float* bo   = (const float*)d_in[10];
    float* out = (float*)d_out;

    u16* ws = (u16*)d_ws;
    const size_t WSZ  = (size_t)CDIM * CDIM;          // 262144
    const size_t TSZ  = (size_t)4 * NTOK * CDIM;      // 8388608
    u16* wqb = ws;
    u16* wkb = wqb + WSZ;
    u16* wvb = wkb + WSZ;
    u16* wob = wvb + WSZ;
    u16* xn  = wob + WSZ;
    u16* qb  = xn + TSZ;
    u16* kbuf= qb + TSZ;
    u16* vtb = kbuf + TSZ;
    u16* aob = vtb + TSZ;

    cvtw_kernel<<<dim3(1024), dim3(256), 0, stream>>>(Wq, wqb, (int)WSZ);
    cvtw_kernel<<<dim3(1024), dim3(256), 0, stream>>>(Wk, wkb, (int)WSZ);
    cvtw_kernel<<<dim3(1024), dim3(256), 0, stream>>>(Wv, wvb, (int)WSZ);
    cvtw_kernel<<<dim3(1024), dim3(256), 0, stream>>>(Wo, wob, (int)WSZ);

    gn_kernel<<<dim3(128), dim3(256), 0, stream>>>(x, gn_w, gn_b, xn);

    float scale = 1.0f / sqrtf((float)CDIM);
    proj_nk_kernel<<<dim3(256, 8, 4), dim3(32), 0, stream>>>(xn, wqb, bq, scale, qb);
    proj_nk_kernel<<<dim3(256, 8, 4), dim3(32), 0, stream>>>(xn, wkb, bk, 1.0f, kbuf);
    proj_v_kernel<<<dim3(32, 64, 4), dim3(32), 0, stream>>>(xn, wvb, bv, vtb);

    flash_kernel<<<dim3(64, 4), dim3(128), 0, stream>>>(qb, kbuf, vtb, aob);

    proj_out_kernel<<<dim3(32, 64, 4), dim3(32), 0, stream>>>(aob, wob, bo, x, out);
}
